// Head_23244363006100
// MI455X (gfx1250) — matrix operand order repaired and checked by reading, not compiled, not run on hardware
//
#include <hip/hip_runtime.h>

typedef __attribute__((ext_vector_type(16))) _Float16     v16h;
typedef __attribute__((ext_vector_type(8)))  float        v8f;
typedef __attribute__((ext_vector_type(4)))  unsigned int v4u;
typedef __attribute__((ext_vector_type(8)))  int          v8i;
typedef __attribute__((ext_vector_type(4)))  int          v4i;

#define EMB 1024
#define HS  128
#define TT  500
#define NB  64
#define MROWS (NB * TT)   // 32000, divisible by 64

static __device__ __forceinline__ int imin(int a, int b) { return a < b ? a : b; }

#if defined(__has_builtin)
#if __has_builtin(__builtin_amdgcn_tensor_load_to_lds)
#define USE_TDM 1
#endif
#endif

#ifdef USE_TDM
// TDM 2D tile load: tile 128(x) x 64(y) elements of 2 bytes from a row-major
// tensor (row pitch 128 elems) into LDS with +16B padding every 256B row
// (-> LDS pitch 272B, matching _Float16 [64][136]).  rows_avail = remaining
// rows in the tensor from the tile start; OOB rows are zero-filled by HW.
static __device__ __forceinline__ void tdm_load_tile_64x128(unsigned lds_byte_addr,
                                                            const void* gsrc,
                                                            int rows_avail) {
  unsigned long long ga = (unsigned long long)(size_t)gsrc;
  v4u g0;
  g0[0] = 1u;                                        // count=1 (valid), user mode
  g0[1] = lds_byte_addr;                             // lds_addr
  g0[2] = (unsigned)ga;                              // global_addr[31:0]
  g0[3] = (unsigned)((ga >> 32) & 0x01FFFFFFu)       // global_addr[56:32]
          | (2u << 30);                              // type = 2 ("image")
  v8i g1;
  g1[0] = (int)((1u << 16)                           // data_size = 2 bytes
              | (1u << 20)                           // pad_enable
              | (5u << 22)                           // pad_interval: 64 DW = 256B
              | (3u << 25));                         // pad_amount: 4 DW = 16B
  g1[1] = (int)(128u << 16);                         // tensor_dim0 = 128 (bits 79:48 lo)
  g1[2] = (int)(((unsigned)rows_avail & 0xFFFFu) << 16); // tensor_dim1 (bits 111:80 lo)
  g1[3] = (int)(128u << 16);                         // tile_dim0 = 128 (bits 127:112)
  g1[4] = 64;                                        // tile_dim1 = 64 (bits 143:128)
  g1[5] = 128;                                       // tensor_dim0_stride = 128
  g1[6] = 0;
  g1[7] = 0;
  v4i z4 = {0, 0, 0, 0};
#if __clang_major__ >= 23
  v8i z8 = {0, 0, 0, 0, 0, 0, 0, 0};
  __builtin_amdgcn_tensor_load_to_lds(g0, g1, z4, z4, z8, 0);
#else
  __builtin_amdgcn_tensor_load_to_lds(g0, g1, z4, z4, 0);
#endif
}

static __device__ __forceinline__ void tdm_wait0() {
#if __has_builtin(__builtin_amdgcn_s_wait_tensorcnt)
  __builtin_amdgcn_s_wait_tensorcnt(0);
#else
  asm volatile("s_wait_tensorcnt 0" ::: "memory");
#endif
}
#endif  // USE_TDM

// ---------------------------------------------------------------------------
// Kernel 1: fused QKV projection.  out[m][n] = sum_c in[m][c] * W[n][c]
// f32 inputs -> f16 q/k/v in workspace (24.6MB, L2-resident for stage 2).
// blockIdx.x = 64-row tile of M=32000, blockIdx.y = 0:q 1:k 2:v
// 256 threads = 8 waves; wave w does rows (w&3)*16, cols (w>>2)*64.
// ---------------------------------------------------------------------------
__launch_bounds__(256)
__global__ void qkv_proj_kernel(const float* __restrict__ x,
                                const float* __restrict__ q_in,
                                const float* __restrict__ Wk,
                                const float* __restrict__ Wq,
                                const float* __restrict__ Wv,
                                _Float16* __restrict__ ws) {
  const int ytype = blockIdx.y;
  const float* in = (ytype == 0) ? q_in : x;
  const float* W  = (ytype == 0) ? Wq : (ytype == 1) ? Wk : Wv;
  _Float16* out = ws + (size_t)ytype * ((size_t)MROWS * HS);

  __shared__ __align__(16) _Float16 Ash[64][40];    // [row][k]  (+pad)
  __shared__ __align__(16) _Float16 Bsh[128][40];   // [n][k] = W rows, no transpose

  const int tid  = threadIdx.x;
  const int lane = tid & 31;
  const int wv   = tid >> 5;
  const int rb   = (wv & 3) * 16;
  const int cb   = (wv >> 2) * 64;
  const int rowbase = blockIdx.x * 64;

  const int m   = lane & 15;                 // A row within wave tile
  const int n   = lane & 15;                 // B/C column within 16-frag
  const int kbA = (lane < 16) ? 0 : 16;      // A K-base (contiguous K, matches B)
  const int kbB = (lane < 16) ? 0 : 16;      // B K-base (contiguous K, matches A)
  const int mr  = (lane < 16) ? 0 : 8;       // C/D row offset for hi lanes

  v8f acc[4] = {};

  for (int kc = 0; kc < EMB; kc += 32) {
    __syncthreads();
    // Stage A tile (coalesced f32 loads, convert to f16)
    #pragma unroll
    for (int it = 0; it < 8; ++it) {
      int e = it * 256 + tid;               // 0..2047
      int r = e >> 5, c = e & 31;
      Ash[r][c] = (_Float16)in[(size_t)(rowbase + r) * EMB + kc + c];
    }
    // Stage B tile row-major [n][k]: straight copy of W rows (coalesced both sides)
    #pragma unroll
    for (int it = 0; it < 16; ++it) {
      int e = it * 256 + tid;               // 0..4095
      int nn = e >> 5, c = e & 31;
      Bsh[nn][c] = (_Float16)W[(size_t)nn * EMB + kc + c];
    }
    __syncthreads();

    v16h a;
    #pragma unroll
    for (int i = 0; i < 8; ++i) a[i]     = Ash[rb + m][kbA + i];
    #pragma unroll
    for (int i = 0; i < 8; ++i) a[8 + i] = Ash[rb + m][kbA + 8 + i];

    #pragma unroll
    for (int f = 0; f < 4; ++f) {
      v16h b;   // contiguous 16 halves -> 2x ds_load_b128
      #pragma unroll
      for (int i = 0; i < 16; ++i) b[i] = Bsh[cb + f * 16 + n][kbB + i];
      acc[f] = __builtin_amdgcn_wmma_f32_16x16x32_f16(
          false, a, false, b, (short)0, acc[f], false, false);
    }
  }

  #pragma unroll
  for (int f = 0; f < 4; ++f) {
    #pragma unroll
    for (int r = 0; r < 8; ++r) {
      int row = rowbase + rb + r + mr;
      int col = cb + f * 16 + n;
      out[(size_t)row * HS + col] = (_Float16)acc[f][r];
    }
  }
}

// ---------------------------------------------------------------------------
// Kernel 2: causal flash attention over f16 q/k/v in workspace.
// blockIdx.y = batch, blockIdx.x = 128-query-row tile; wave owns 16 q-rows.
// K/V 64x128 tiles fetched by the Tensor Data Mover (TENSORcnt) when available.
// ---------------------------------------------------------------------------
__launch_bounds__(256)
__global__ void flash_attn_kernel(const _Float16* __restrict__ ws,
                                  float* __restrict__ out) {
  const int b    = blockIdx.y;
  const int q0   = blockIdx.x * 128;
  const int tid  = threadIdx.x;
  const int lane = tid & 31;
  const int wv   = tid >> 5;

  const _Float16* Q = ws + (size_t)b * TT * HS;
  const _Float16* K = ws + (size_t)MROWS * HS + (size_t)b * TT * HS;
  const _Float16* V = ws + 2 * (size_t)MROWS * HS + (size_t)b * TT * HS;

  __shared__ __align__(16) _Float16 Ksh[64][136];     // pitch 272B (TDM-padded)
  __shared__ __align__(16) _Float16 Vsh[64][136];
  __shared__ __align__(16) _Float16 Psh[8][16][72];   // per-wave P staging

  const int m   = lane & 15;
  const int n   = lane & 15;
  const int kbA = (lane < 16) ? 0 : 16;
  const int kbB = (lane < 16) ? 0 : 16;
  const int mr  = (lane < 16) ? 0 : 8;

  const int qrow_w  = q0 + wv * 16;              // wave's first query row
  const int tq_load = imin(qrow_w + m, TT - 1);  // clamped for padded rows

  // Load the wave's Q rows once as 4 A-fragments (K = 0..127 in chunks of 32)
  v16h qa[4];
  #pragma unroll
  for (int kc = 0; kc < 4; ++kc) {
    #pragma unroll
    for (int i = 0; i < 8; ++i) qa[kc][i]     = Q[(size_t)tq_load * HS + kc * 32 + kbA + i];
    #pragma unroll
    for (int i = 0; i < 8; ++i) qa[kc][8 + i] = Q[(size_t)tq_load * HS + kc * 32 + kbA + 8 + i];
  }

  float mrun[8], lrun[8];
  v8f oacc[8] = {};
  #pragma unroll
  for (int r = 0; r < 8; ++r) { mrun[r] = -1e30f; lrun[r] = 0.0f; }

  const int kmax   = imin(TT, q0 + 128);        // causal: no keys beyond q-tile end
  const int ntiles = (kmax + 63) >> 6;

  for (int kt = 0; kt < ntiles; ++kt) {
    const int keybase = kt * 64;
    __syncthreads();   // previous tile fully consumed before overwrite
#ifdef USE_TDM
    if (wv == 0) {
      tdm_load_tile_64x128((unsigned)(size_t)&Ksh[0][0],
                           K + (size_t)keybase * HS, TT - keybase);
      tdm_load_tile_64x128((unsigned)(size_t)&Vsh[0][0],
                           V + (size_t)keybase * HS, TT - keybase);
      tdm_wait0();
    }
#else
    #pragma unroll
    for (int it = 0; it < 4; ++it) {
      int idx = it * 256 + tid;                 // 0..1023 uint4 slots
      int key = idx >> 4;
      int col = (idx & 15) * 8;
      int krow = imin(keybase + key, TT - 1);
      *(uint4*)&Ksh[key][col] = *(const uint4*)&K[(size_t)krow * HS + col];
      *(uint4*)&Vsh[key][col] = *(const uint4*)&V[(size_t)krow * HS + col];
    }
#endif
    __syncthreads();

    // S = Q * K^T for this wave's 16 rows x 64 keys
    v8f s[4] = {};
    #pragma unroll
    for (int kc = 0; kc < 4; ++kc) {
      #pragma unroll
      for (int f = 0; f < 4; ++f) {
        v16h bb;  // contiguous -> 2x ds_load_b128
        #pragma unroll
        for (int i = 0; i < 16; ++i) bb[i] = Ksh[f * 16 + n][kc * 32 + kbB + i];
        s[f] = __builtin_amdgcn_wmma_f32_16x16x32_f16(
            false, qa[kc], false, bb, (short)0, s[f], false, false);
      }
    }

    // Causal + tail mask
    #pragma unroll
    for (int f = 0; f < 4; ++f) {
      int col = keybase + f * 16 + n;
      #pragma unroll
      for (int r = 0; r < 8; ++r) {
        int tq = qrow_w + r + mr;
        if (col > tq || col >= TT) s[f][r] = -1e30f;
      }
    }

    // Online softmax: row reductions across the 16-lane half-wave
    float pscale[8];
    #pragma unroll
    for (int r = 0; r < 8; ++r) {
      float mx = fmaxf(fmaxf(s[0][r], s[1][r]), fmaxf(s[2][r], s[3][r]));
      #pragma unroll
      for (int off = 8; off >= 1; off >>= 1) mx = fmaxf(mx, __shfl_xor(mx, off, 32));
      float mnew = fmaxf(mrun[r], mx);
      float sc   = __expf(mrun[r] - mnew);
      float rsum = 0.0f;
      #pragma unroll
      for (int f = 0; f < 4; ++f) {
        float p = __expf(s[f][r] - mnew);
        s[f][r] = p;
        rsum += p;
      }
      #pragma unroll
      for (int off = 8; off >= 1; off >>= 1) rsum += __shfl_xor(rsum, off, 32);
      lrun[r]   = lrun[r] * sc + rsum;
      mrun[r]   = mnew;
      pscale[r] = sc;
    }

    // Rescale running output
    #pragma unroll
    for (int f = 0; f < 8; ++f)
      #pragma unroll
      for (int r = 0; r < 8; ++r) oacc[f][r] *= pscale[r];

    // P (C-layout) -> LDS -> A-layout fragments
    #pragma unroll
    for (int f = 0; f < 4; ++f)
      #pragma unroll
      for (int r = 0; r < 8; ++r)
        Psh[wv][r + mr][f * 16 + n] = (_Float16)s[f][r];
    asm volatile("" ::: "memory");   // same-wave DS ops are in-order; stop reordering

    // O += P * V   (K = 64 keys in two chunks of 32)
    #pragma unroll
    for (int kc = 0; kc < 2; ++kc) {
      v16h pa;
      #pragma unroll
      for (int i = 0; i < 8; ++i) pa[i]     = Psh[wv][m][kc * 32 + kbA + i];
      #pragma unroll
      for (int i = 0; i < 8; ++i) pa[8 + i] = Psh[wv][m][kc * 32 + kbA + 8 + i];
      #pragma unroll
      for (int f = 0; f < 8; ++f) {
        v16h bv;
        #pragma unroll
        for (int i = 0; i < 16; ++i) bv[i] = Vsh[kc * 32 + kbB + i][f * 16 + n];
        oacc[f] = __builtin_amdgcn_wmma_f32_16x16x32_f16(
            false, pa, false, bv, (short)0, oacc[f], false, false);
      }
    }
  }

  // Normalize and store f32 output [B,T,H]
  #pragma unroll
  for (int r = 0; r < 8; ++r) {
    int tq = qrow_w + r + mr;
    if (tq < TT) {
      float inv = 1.0f / lrun[r];
      #pragma unroll
      for (int f = 0; f < 8; ++f)
        out[((size_t)b * TT + tq) * HS + f * 16 + n] = oacc[f][r] * inv;
    }
  }
}

extern "C" void kernel_launch(void* const* d_in, const int* in_sizes, int n_in,
                              void* d_out, int out_size, void* d_ws, size_t ws_size,
                              hipStream_t stream) {
  const float* x    = (const float*)d_in[0];
  const float* q_in = (const float*)d_in[1];
  const float* Wk   = (const float*)d_in[2];
  const float* Wq   = (const float*)d_in[3];
  const float* Wv   = (const float*)d_in[4];
  _Float16* ws = (_Float16*)d_ws;      // needs 3*32000*128*2 = 24.6 MB
  float* out = (float*)d_out;

  dim3 g1(MROWS / 64, 3);
  qkv_proj_kernel<<<g1, 256, 0, stream>>>(x, q_in, Wk, Wq, Wv, ws);

  dim3 g2((TT + 127) / 128, NB);
  flash_attn_kernel<<<g2, 256, 0, stream>>>(ws, out);
}
